// Attribute_visual_atten_32856499814690
// MI455X (gfx1250) — matrix operand order repaired and checked by reading, not compiled, not run on hardware
//
#include <hip/hip_runtime.h>

typedef __attribute__((ext_vector_type(16))) __bf16 v16bf;
typedef __attribute__((ext_vector_type(8)))  float  v8f;

union FragBF { v16bf v; unsigned int u[8]; };

// truncating f32 -> bf16 (1 VALU op)
__device__ __forceinline__ unsigned short f2bf(float x) {
    return (unsigned short)(__builtin_bit_cast(unsigned int, x) >> 16);
}
// pack two f32 -> bf16x2 in one v_perm_b32: dword = { bf(y), bf(x) }
__device__ __forceinline__ unsigned int pack_bf(float x, float y) {
    return __builtin_amdgcn_perm(__builtin_bit_cast(unsigned int, y),
                                 __builtin_bit_cast(unsigned int, x),
                                 0x07060302u);
}

// ---------------- Kernel 1: q = v @ W_alpha  ->  bf16, padded to 320 rows ----
__global__ __launch_bounds__(256)
void qproj_kernel(const float* __restrict__ v,
                  const float* __restrict__ W,
                  unsigned short* __restrict__ qw) {
    int gid = blockIdx.x * 256 + threadIdx.x;
    if (gid >= 320 * 2048) return;
    int i = gid >> 11;
    int f = gid & 2047;
    float acc = 0.f;
    if (i < 312) {
        const float* vr = v + i * 300;
        #pragma unroll 4
        for (int d = 0; d < 300; ++d)
            acc += vr[d] * W[d * 2048 + f];
    }
    qw[gid] = f2bf(acc);
}

// ---------------- Kernel 2: fused scores -> softmax -> attend ----------------
// grid: (10 i-tiles of 32 rows, 64 batches), 256 threads (8 waves)
__global__ __launch_bounds__(256)
void atten_kernel(const float* __restrict__ vf,          // [64][2048][196]
                  const unsigned short* __restrict__ qw, // [320][2048] bf16
                  float* __restrict__ out) {             // [64][312][2048]
    constexpr int F = 2048, R = 196, I = 312;
    constexpr int CH = 38;    // chunk row stride (halfwords): dword aligned, conflict-free
    constexpr int SS = 260;   // scores row stride (floats): conflict-free in softmax
    constexpr int AS = 228;   // atten row stride (halfwords)

    __shared__ __align__(16) unsigned short chunk[256 * CH];  // vf slab^T, bf16 (16 r-tiles)
    __shared__ __align__(16) float          scores[32 * SS];
    __shared__ __align__(16) unsigned short atten[32 * AS];
    __shared__ float red[32 * 8];

    const int tid  = threadIdx.x;
    const int wave = tid >> 5;
    const int lane = tid & 31;
    const int ln   = lane & 15;
    const int hi   = lane >> 4;
    const int itile = blockIdx.x;
    const int b     = blockIdx.y;
    const int ibase = itile * 32;

    const float* vfb = vf + (size_t)b * F * R;
    const unsigned int* q32 = (const unsigned int*)qw;

    // zero-pad chunk rows r = 196..255 once (persist across K steps)
    for (int idx = tid; idx < 60 * 32; idx += 256) {
        int r = 196 + (idx >> 5), k = idx & 31;
        chunk[r * CH + k] = 0;
    }

    // Fixed, uniform tile ownership: mt = wave>>2, nt = (wave&3) + 4j, j=0..3
    const int mt1  = wave >> 2;
    const int ntb  = wave & 3;
    const int arow = ibase + mt1 * 16 + ln;      // A rows padded to 320 in qw

    // -------- Phase 1: scores[32,256pad] = q_tile[32,2048] @ vf_b[2048,196] --
    v8f acc0 = {}, acc1 = {}, acc2 = {}, acc3 = {};
    for (int ks = 0; ks < F / 32; ++ks) {
        const int k0 = ks * 32;
        __syncthreads();
        // cooperative load: vf[k0+k..k0+k+1][r] -> one packed dword in chunk[r][k..k+1]
        for (int idx = tid; idx < 16 * R; idx += 256) {
            int kp2 = idx / R, r = idx - kp2 * R;
            int k = kp2 * 2;
            float x0 = vfb[(size_t)(k0 + k) * R + r];
            float x1 = vfb[(size_t)(k0 + k + 1) * R + r];
            *(unsigned int*)&chunk[r * CH + k] = pack_bf(x0, x1);
        }
        __syncthreads();

        // A fragment (16x32 bf16, k = 16*hi + s) for this wave's m-tile
        FragBF af;
        #pragma unroll
        for (int vv = 0; vv < 8; ++vv) {
            int kp = (hi << 4) + (vv << 1);
            af.u[vv] = q32[arow * 1024 + ((k0 + kp) >> 1)];
        }
        const unsigned int* cp = (const unsigned int*)chunk;
        FragBF b0, b1, b2, b3;
        const int rbase = ntb * 16 + ln;         // + 64*j per tile
        #pragma unroll
        for (int vv = 0; vv < 8; ++vv) {         // B layout: k = hi*16 + 2*vv
            int o = hi * 8 + vv;
            b0.u[vv] = cp[(rbase      ) * (CH / 2) + o];
            b1.u[vv] = cp[(rbase +  64) * (CH / 2) + o];
            b2.u[vv] = cp[(rbase + 128) * (CH / 2) + o];
            b3.u[vv] = cp[(rbase + 192) * (CH / 2) + o];
        }
        acc0 = __builtin_amdgcn_wmma_f32_16x16x32_bf16(false, af.v, false, b0.v, (short)0, acc0, false, false);
        acc1 = __builtin_amdgcn_wmma_f32_16x16x32_bf16(false, af.v, false, b1.v, (short)0, acc1, false, false);
        acc2 = __builtin_amdgcn_wmma_f32_16x16x32_bf16(false, af.v, false, b2.v, (short)0, acc2, false, false);
        acc3 = __builtin_amdgcn_wmma_f32_16x16x32_bf16(false, af.v, false, b3.v, (short)0, acc3, false, false);
    }
    __syncthreads();
    // spill accumulators (C/D layout: VGPR p -> M = p + 8*hi) into LDS scores
    {
        int m0 = mt1 * 16 + hi * 8;
        int c0 = ntb * 16 + ln;
        #pragma unroll
        for (int p = 0; p < 8; ++p) {
            float* srow = scores + (m0 + p) * SS + c0;
            srow[  0] = acc0[p];
            srow[ 64] = acc1[p];
            srow[128] = acc2[p];
            srow[192] = acc3[p];
        }
    }
    __syncthreads();

    // -------- Softmax over r (8 lanes per row; all 8 within one wave) --------
    {
        int row = tid >> 3, sub = tid & 7;
        float* srow = scores + row * SS;
        float mx = -3.4e38f;
        for (int c = sub; c < R; c += 8) mx = fmaxf(mx, srow[c]);
        red[row * 8 + sub] = mx;
        float rm = -3.4e38f;
        #pragma unroll
        for (int jj = 0; jj < 8; ++jj) rm = fmaxf(rm, red[row * 8 + jj]);
        float s = 0.f;
        for (int c = sub; c < R; c += 8) {
            float e = __expf(srow[c] - rm);
            srow[c] = e;
            s += e;
        }
        red[row * 8 + sub] = s;
        float rs = 0.f;
        #pragma unroll
        for (int jj = 0; jj < 8; ++jj) rs += red[row * 8 + jj];
        float inv = 1.0f / rs;
        for (int c = sub; c < R; c += 8)
            atten[row * AS + c] = f2bf(srow[c] * inv);
        for (int c = R + sub; c < AS; c += 8)   // zero K-padding (196..227)
            atten[row * AS + c] = 0;
    }
    __syncthreads();

    // -------- Phase 2: out[32,2048] = atten[32,224] @ vf_b^T[224,2048] -------
    // Hoist A frags for BOTH m-tiles (wave-invariant); each B frag feeds 2 WMMAs,
    // halving global fp32 traffic and bf16 pack work.
    const unsigned int* ap = (const unsigned int*)atten;
    const bool fullTile = (ibase + 32 <= I);     // uniform: only last i-tile guarded
    FragBF afr0[7], afr1[7];
    #pragma unroll
    for (int ks = 0; ks < 7; ++ks) {
        #pragma unroll
        for (int vv = 0; vv < 8; ++vv) {
            int kp = (hi << 4) + (vv << 1);
            int ci = (ks * 32 + kp) >> 1;
            afr0[ks].u[vv] = ap[ ln       * (AS / 2) + ci];
            afr1[ks].u[vv] = ap[(ln + 16) * (AS / 2) + ci];
        }
    }
    for (int nt = wave; nt < 128; nt += 8) {
        int fbase = nt * 16;
        const float* col = vfb + (size_t)(fbase + ln) * R; // r-contiguous
        v8f accD0 = {}, accD1 = {};
        #pragma unroll
        for (int ks = 0; ks < 7; ++ks) {
            FragBF bfr;
            int rb = ks * 32 + hi * 16;
            if (ks < 6) {                    // always in-bounds (rb+15 <= 191)
                #pragma unroll
                for (int vv = 0; vv < 8; ++vv) {
                    float2 p2 = *(const float2*)(col + rb + 2 * vv);
                    bfr.u[vv] = pack_bf(p2.x, p2.y);
                }
            } else {                         // tail K-step: guard r < 196
                #pragma unroll
                for (int vv = 0; vv < 8; ++vv) {
                    int r0 = rb + 2 * vv;
                    float x0 = (r0     < R) ? col[r0]     : 0.f;
                    float x1 = (r0 + 1 < R) ? col[r0 + 1] : 0.f;
                    bfr.u[vv] = pack_bf(x0, x1);
                }
            }
            accD0 = __builtin_amdgcn_wmma_f32_16x16x32_bf16(
                false, afr0[ks].v, false, bfr.v, (short)0, accD0, false, false);
            accD1 = __builtin_amdgcn_wmma_f32_16x16x32_bf16(
                false, afr1[ks].v, false, bfr.v, (short)0, accD1, false, false);
        }
        int m0 = hi * 8;
        float* orow0 = out + ((size_t)b * I + (ibase + m0     )) * F + fbase + ln;
        float* orow1 = out + ((size_t)b * I + (ibase + m0 + 16)) * F + fbase + ln;
        if (fullTile) {
            #pragma unroll
            for (int p = 0; p < 8; ++p) {
                orow0[(size_t)p * F] = accD0[p];
                orow1[(size_t)p * F] = accD1[p];
            }
        } else {
            #pragma unroll
            for (int p = 0; p < 8; ++p) {
                if (ibase + m0 + p      < I) orow0[(size_t)p * F] = accD0[p];
                if (ibase + m0 + p + 16 < I) orow1[(size_t)p * F] = accD1[p];
            }
        }
    }
}

extern "C" void kernel_launch(void* const* d_in, const int* in_sizes, int n_in,
                              void* d_out, int out_size, void* d_ws, size_t ws_size,
                              hipStream_t stream) {
    (void)in_sizes; (void)n_in; (void)out_size; (void)ws_size;
    const float* vf = (const float*)d_in[0];   // [64,2048,196]
    const float* v  = (const float*)d_in[1];   // [312,300]
    const float* W  = (const float*)d_in[2];   // [300,2048]
    float* out = (float*)d_out;                // [64,312,2048]
    unsigned short* qw = (unsigned short*)d_ws; // 320*2048 bf16 = 1.25 MB

    qproj_kernel<<<(320 * 2048) / 256, 256, 0, stream>>>(v, W, qw);
    atten_kernel<<<dim3(10, 64), 256, 0, stream>>>(vf, qw, out);
}
